// GAT_65377992180227
// MI455X (gfx1250) — matrix operand order repaired and checked by reading, not compiled, not run on hardware
//
#include <hip/hip_runtime.h>
#include <hip/hip_bf16.h>

// ---------------------------------------------------------------------------
// GAT-style graph transformer for MI455X (gfx1250, wave32, WMMA).
// All matmuls run through v_wmma_f32_16x16x32_bf16 (bf16 in / f32 accum).
// GEMM tiles are double-buffered via global_load_async_to_lds_b128 (ASYNCcnt).
// ---------------------------------------------------------------------------

#define B_    8
#define N_    1024
#define D_    256
#define H_    8
#define HD_   32
#define DEPTH_ 3
#define EPS_  1e-5f

typedef __bf16 bf16;
typedef __attribute__((ext_vector_type(8)))  bf16  bf16x8;
typedef __attribute__((ext_vector_type(16))) bf16  bf16x16;
typedef __attribute__((ext_vector_type(8)))  float f32x8;

__device__ __forceinline__ f32x8 wmma_bf16(bf16x16 a, bf16x16 b, f32x8 c) {
  return __builtin_amdgcn_wmma_f32_16x16x32_bf16(false, a, false, b,
                                                 (short)0, c, false, false);
}

__device__ __forceinline__ bf16x16 join8(bf16x8 lo, bf16x8 hi) {
  return __builtin_shufflevector(lo, hi, 0,1,2,3,4,5,6,7,8,9,10,11,12,13,14,15);
}

// Async global -> LDS copy of 16 bytes (per-lane addresses, ASYNCcnt).
// LDS byte address = low 32 bits of the generic __shared__ pointer.
__device__ __forceinline__ void async_g2l_b128(const void* lds_ptr,
                                               const void* gptr) {
  unsigned l = (unsigned)(unsigned long long)(uintptr_t)lds_ptr;
  unsigned long long g = (unsigned long long)(uintptr_t)gptr;
  asm volatile("global_load_async_to_lds_b128 %0, %1, off"
               :: "v"(l), "v"(g) : "memory");
}
__device__ __forceinline__ void wait_async0() {
  asm volatile("s_wait_asynccnt 0x0" ::: "memory");
}

__device__ __forceinline__ float rmax16(float x) {
  x = fmaxf(x, __shfl_xor(x, 1));
  x = fmaxf(x, __shfl_xor(x, 2));
  x = fmaxf(x, __shfl_xor(x, 4));
  x = fmaxf(x, __shfl_xor(x, 8));
  return x;
}
__device__ __forceinline__ float rsum16(float x) {
  x += __shfl_xor(x, 1); x += __shfl_xor(x, 2);
  x += __shfl_xor(x, 4); x += __shfl_xor(x, 8);
  return x;
}

// ---------------------------------------------------------------------------
// Generic bf16 WMMA GEMM:  C[M,N] = A[M,K] @ B[K,N] (+ bias[N])
// 256 threads = 8 waves; block tile 128x64x32; wave tile 32x32 (2x2 WMMA).
// Double-buffered async global->LDS staging: next tile's loads overlap the
// current tile's WMMAs; one barrier per k-step. blockIdx.z batches by stride.
// ---------------------------------------------------------------------------
#define BM 128
#define BN 64
#define BK 32

template<bool HAS_BIAS, bool WRITE_BF>
__global__ __launch_bounds__(256) void k_gemm_bf16(
    const bf16* __restrict__ Ag, const bf16* __restrict__ Bg,
    float* __restrict__ Cg, bf16* __restrict__ Cbf,
    const float* __restrict__ bias,
    int Nn, int K,
    long long sA, long long sB, long long sC)
{
  __shared__ bf16 As[2][BM][BK + 8];  // +8 keeps 16B alignment, spreads banks
  __shared__ bf16 Bs[2][BK][BN + 8];

  const int bz = blockIdx.z;
  Ag += (long long)bz * sA;
  Bg += (long long)bz * sB;
  const long long cbase = (long long)bz * sC;

  const int tid  = threadIdx.x;
  const int lane = tid & 31;
  const int wave = tid >> 5;
  const int wm = wave >> 1, wn = wave & 1;     // 4 x 2 wave grid
  const int half = lane >> 4, r = lane & 15;
  const int bm0 = blockIdx.y * BM, bn0 = blockIdx.x * BN;

  // Per-thread staging coordinates (fixed across k-steps).
  const int arow0 = tid >> 2;               // A chunk 0: rows 0..63
  const int ac8   = (tid & 3) << 3;
  const int arow1 = arow0 + 64;             // A chunk 1: rows 64..127
  const int brow  = tid >> 3;               // B: 32 rows x 8 chunks
  const int bc8   = (tid & 7) << 3;

  f32x8 z = {0.f,0.f,0.f,0.f,0.f,0.f,0.f,0.f};
  f32x8 acc[2][2] = {{z, z}, {z, z}};

  const int nk = K / BK;

  // Prologue: stage tile 0 into buffer 0.
  async_g2l_b128(&As[0][arow0][ac8],
                 Ag + (long long)(bm0 + arow0) * K + ac8);
  async_g2l_b128(&As[0][arow1][ac8],
                 Ag + (long long)(bm0 + arow1) * K + ac8);
  async_g2l_b128(&Bs[0][brow][bc8],
                 Bg + (long long)brow * Nn + bn0 + bc8);

  for (int kk = 0; kk < nk; ++kk) {
    wait_async0();        // tile kk resident (its loads flew during compute kk-1)
    __syncthreads();      // all waves' async writes visible

    if (kk + 1 < nk) {    // stage tile kk+1 into the other buffer
      const int nb = (kk + 1) & 1;
      const int k1 = (kk + 1) * BK;
      async_g2l_b128(&As[nb][arow0][ac8],
                     Ag + (long long)(bm0 + arow0) * K + k1 + ac8);
      async_g2l_b128(&As[nb][arow1][ac8],
                     Ag + (long long)(bm0 + arow1) * K + k1 + ac8);
      async_g2l_b128(&Bs[nb][brow][bc8],
                     Bg + (long long)(k1 + brow) * Nn + bn0 + bc8);
    }

    const int cb = kk & 1;
    bf16x16 af[2], bfr[2];
#pragma unroll
    for (int i = 0; i < 2; ++i) {
      const bf16* base = &As[cb][wm * 32 + i * 16 + r][0];
      af[i] = join8(*(const bf16x8*)(base + half * 8),
                    *(const bf16x8*)(base + 16 + half * 8));
    }
#pragma unroll
    for (int j = 0; j < 2; ++j) {
      const bf16* base = &Bs[cb][0][wn * 32 + j * 16 + r];
      for (int s = 0; s < 16; ++s)
        bfr[j][s] = base[((s & 7) + half * 8 + (s >> 3) * 16) * (BN + 8)];
    }
#pragma unroll
    for (int i = 0; i < 2; ++i)
#pragma unroll
      for (int j = 0; j < 2; ++j)
        acc[i][j] = wmma_bf16(af[i], bfr[j], acc[i][j]);
  }

#pragma unroll
  for (int i = 0; i < 2; ++i)
#pragma unroll
    for (int j = 0; j < 2; ++j) {
      const int col = bn0 + wn * 32 + j * 16 + r;
      const float bv = HAS_BIAS ? bias[col] : 0.f;
#pragma unroll
      for (int p = 0; p < 8; ++p) {
        const int rowg = bm0 + wm * 32 + i * 16 + p + 8 * half;
        const float v = acc[i][j][p] + bv;
        const long long idx = cbase + (long long)rowg * Nn + col;
        Cg[idx] = v;
        if (WRITE_BF) Cbf[idx] = (bf16)v;
      }
    }
}

// ---------------------------------------------------------------------------
// Attention pass 1: masked-softmax row stats (max, sumexp) per (b,h,row).
// One wave handles 16 rows; HD=32 => one WMMA per 16x16 score tile.
// ---------------------------------------------------------------------------
__global__ __launch_bounds__(32) void k_attn_stats(
    const bf16* __restrict__ Q,   // [B,H,N,HD]
    const bf16* __restrict__ Kt,  // [B,H,HD,N]
    const float* __restrict__ adj,
    float* __restrict__ rowM, float* __restrict__ rowS)  // [B,H,N]
{
  const int i0 = blockIdx.x << 4;
  const int h = blockIdx.y, b = blockIdx.z;
  const int lane = threadIdx.x;
  const int half = lane >> 4, r = lane & 15;
  const long long bh = (long long)(b * H_ + h);
  const bf16* Qp = Q + (bh * N_ + i0) * HD_;
  const bf16* Kp = Kt + bh * HD_ * N_;

  bf16x16 aq = join8(*(const bf16x8*)(Qp + r * HD_ + half * 8),
                     *(const bf16x8*)(Qp + r * HD_ + 16 + half * 8));
  const float scale = 0.17677669529663687f;  // 1/sqrt(32)
  const f32x8 z8 = {0.f,0.f,0.f,0.f,0.f,0.f,0.f,0.f};

  float m[8], s[8];
#pragma unroll
  for (int p = 0; p < 8; ++p) { m[p] = -3.0e38f; s[p] = 0.f; }

  for (int j0 = 0; j0 < N_; j0 += 16) {
    const bf16* kb = Kp + j0 + r;
    bf16x16 bk;
    for (int s = 0; s < 16; ++s)
      bk[s] = kb[(long long)((s & 7) + half * 8 + (s >> 3) * 16) * N_];
    f32x8 c = wmma_bf16(aq, bk, z8);
    const int j = j0 + r;
#pragma unroll
    for (int p = 0; p < 8; ++p) {
      const int i = i0 + p + 8 * half;
      float x = (adj[(long long)i * N_ + j] > 0.f) ? c[p] * scale : -1e9f;
      float nm = fmaxf(m[p], rmax16(x));
      float ts = rsum16(__expf(x - nm));
      s[p] = s[p] * __expf(m[p] - nm) + ts;
      m[p] = nm;
    }
  }
  if (r == 0) {
#pragma unroll
    for (int p = 0; p < 8; ++p) {
      const long long idx = bh * N_ + i0 + p + 8 * half;
      rowM[idx] = m[p];
      rowS[idx] = s[p];
    }
  }
}

// ---------------------------------------------------------------------------
// Attention pass 2: P = softmax(S), accumulate head-averaged a_mean
// (global f32 atomics) and O = P @ V (WMMA, P re-fragmented through LDS).
// ---------------------------------------------------------------------------
__global__ __launch_bounds__(32) void k_attn_apply(
    const bf16* __restrict__ Q, const bf16* __restrict__ Kt,
    const bf16* __restrict__ V, const float* __restrict__ adj,
    const float* __restrict__ rowM, const float* __restrict__ rowS,
    float* __restrict__ amean,  // [B,N,N], pre-zeroed
    float* __restrict__ O)      // [B,N,D]
{
  __shared__ bf16 P[16][40];
  const int i0 = blockIdx.x << 4;
  const int h = blockIdx.y, b = blockIdx.z;
  const int lane = threadIdx.x;
  const int half = lane >> 4, r = lane & 15;
  const long long bh = (long long)(b * H_ + h);
  const bf16* Qp = Q + (bh * N_ + i0) * HD_;
  const bf16* Kp = Kt + bh * HD_ * N_;
  const bf16* Vp = V + bh * N_ * HD_;

  bf16x16 aq = join8(*(const bf16x8*)(Qp + r * HD_ + half * 8),
                     *(const bf16x8*)(Qp + r * HD_ + 16 + half * 8));
  const float scale = 0.17677669529663687f;
  const f32x8 z8 = {0.f,0.f,0.f,0.f,0.f,0.f,0.f,0.f};

  float mrow[8], zinv[8];
#pragma unroll
  for (int p = 0; p < 8; ++p) {
    const long long idx = bh * N_ + i0 + p + 8 * half;
    mrow[p] = rowM[idx];
    float zs = rowS[idx];
    zinv[p] = (zs > 0.f) ? 1.f / zs : 0.f;
  }

  f32x8 oacc[2] = {z8, z8};

  for (int j0 = 0; j0 < N_; j0 += 32) {
#pragma unroll
    for (int jt = 0; jt < 2; ++jt) {
      const bf16* kb = Kp + j0 + jt * 16 + r;
      bf16x16 bk;
      for (int s = 0; s < 16; ++s)
        bk[s] = kb[(long long)((s & 7) + half * 8 + (s >> 3) * 16) * N_];
      f32x8 c = wmma_bf16(aq, bk, z8);
      const int j = j0 + jt * 16 + r;
#pragma unroll
      for (int p = 0; p < 8; ++p) {
        const int i = i0 + p + 8 * half;
        float pn = 0.f;
        if (adj[(long long)i * N_ + j] > 0.f)
          pn = __expf(c[p] * scale - mrow[p]) * zinv[p];
        atomicAdd(&amean[((long long)b * N_ + i) * N_ + j], pn * 0.125f);
        P[p + 8 * half][jt * 16 + r] = (bf16)pn;
      }
    }
    __syncthreads();
    bf16x16 ap = join8(*(const bf16x8*)&P[r][half * 8],
                       *(const bf16x8*)&P[r][16 + half * 8]);
#pragma unroll
    for (int dt = 0; dt < 2; ++dt) {
      const bf16* vb = Vp + (long long)j0 * HD_ + dt * 16 + r;
      bf16x16 bv;
      for (int s = 0; s < 16; ++s)
        bv[s] = vb[((s & 7) + half * 8 + (s >> 3) * 16) * HD_];
      oacc[dt] = wmma_bf16(ap, bv, oacc[dt]);
    }
    __syncthreads();
  }

#pragma unroll
  for (int dt = 0; dt < 2; ++dt)
#pragma unroll
    for (int p = 0; p < 8; ++p) {
      const int i = i0 + p + 8 * half;
      O[((long long)b * N_ + i) * D_ + h * HD_ + dt * 16 + r] = oacc[dt][p];
    }
}

// ---------------------------------------------------------------------------
// h = LayerNorm(h + y) * gamma + beta ; also emit bf16 copy. 1 wave / row.
// ---------------------------------------------------------------------------
__global__ __launch_bounds__(32) void k_residual_ln(
    const float* __restrict__ hin, const float* __restrict__ y,
    const float* __restrict__ gamma, const float* __restrict__ beta,
    float* __restrict__ hout, bf16* __restrict__ hbf)
{
  const long long row = blockIdx.x;
  const int lane = threadIdx.x;
  const float* hp = hin + row * D_;
  const float* yp = y + row * D_;
  float v[8];
  float sum = 0.f;
#pragma unroll
  for (int t = 0; t < 8; ++t) {
    v[t] = hp[lane + t * 32] + yp[lane + t * 32];
    sum += v[t];
  }
#pragma unroll
  for (int o = 16; o >= 1; o >>= 1) sum += __shfl_xor(sum, o);
  const float mu = sum * (1.f / D_);
  float sq = 0.f;
#pragma unroll
  for (int t = 0; t < 8; ++t) { float d = v[t] - mu; sq += d * d; }
#pragma unroll
  for (int o = 16; o >= 1; o >>= 1) sq += __shfl_xor(sq, o);
  const float rstd = rsqrtf(sq * (1.f / D_) + EPS_);
#pragma unroll
  for (int t = 0; t < 8; ++t) {
    const int c = lane + t * 32;
    float outv = (v[t] - mu) * rstd * gamma[c] + beta[c];
    hout[row * D_ + c] = outv;
    hbf[row * D_ + c]  = (bf16)outv;
  }
}

// ---------------------------------------------------------------------------
// Small helpers: converts / transposes / zero-fill
// ---------------------------------------------------------------------------
__global__ void k_f32_to_bf16(const float* __restrict__ src,
                              bf16* __restrict__ dst, long long n) {
  long long i = (long long)blockIdx.x * blockDim.x + threadIdx.x;
  if (i < n) dst[i] = (bf16)src[i];
}

__global__ void k_transpose_bf16(const float* __restrict__ W,
                                 bf16* __restrict__ Wt, int R, int C) {
  long long i = (long long)blockIdx.x * blockDim.x + threadIdx.x;
  if (i < (long long)R * C) {
    int rr = (int)(i / C), cc = (int)(i % C);
    Wt[(long long)cc * R + rr] = (bf16)W[i];
  }
}

// src [B,N,H,HD] f32 -> dst bf16 [B,H,N,HD] (TRANS=0) or [B,H,HD,N] (TRANS=1)
template<int TRANS>
__global__ void k_qkv_convert(const float* __restrict__ src,
                              bf16* __restrict__ dst) {
  long long i = (long long)blockIdx.x * blockDim.x + threadIdx.x;
  if (i >= (long long)B_ * N_ * D_) return;
  const int d = (int)(i & (HD_ - 1));
  const int h = (int)((i >> 5) & (H_ - 1));
  const int n = (int)((i >> 8) & (N_ - 1));
  const int b = (int)(i >> 18);
  const long long bh = (long long)(b * H_ + h);
  const bf16 val = (bf16)src[i];
  if (TRANS) dst[(bh * HD_ + d) * N_ + n] = val;
  else       dst[(bh * N_ + n) * HD_ + d] = val;
}

__global__ void k_fill0(float* __restrict__ p, long long n) {
  long long i = (long long)blockIdx.x * blockDim.x + threadIdx.x;
  if (i < n) p[i] = 0.f;
}

// ---------------------------------------------------------------------------
extern "C" void kernel_launch(void* const* d_in, const int* in_sizes, int n_in,
                              void* d_out, int out_size, void* d_ws, size_t ws_size,
                              hipStream_t stream)
{
  const float* x     = (const float*)d_in[0];
  const float* adj   = (const float*)d_in[1];
  const float* W0    = (const float*)d_in[2];
  const float* Wq    = (const float*)d_in[3];
  const float* bq    = (const float*)d_in[4];
  const float* Wk    = (const float*)d_in[5];
  const float* bk    = (const float*)d_in[6];
  const float* Wv    = (const float*)d_in[7];
  const float* bv    = (const float*)d_in[8];
  const float* Wo    = (const float*)d_in[9];
  const float* bo    = (const float*)d_in[10];
  const float* gamma = (const float*)d_in[11];
  const float* beta  = (const float*)d_in[12];

  const long long BND = (long long)B_ * N_ * D_;   // 2,097,152
  const long long BNN = (long long)B_ * N_ * N_;   // 8,388,608
  const long long BHN = (long long)B_ * H_ * N_;
  const long long DD  = (long long)D_ * D_;

  char* w = (char*)d_ws;
  auto take = [&](long long bytes) -> void* {
    void* p = (void*)w;
    w += (bytes + 255) & ~255LL;
    return p;
  };
  float* hbuf  = (float*)take(BND * 4);
  float* tmp   = (float*)take(BND * 4);   // q/k/v projection out, then y
  float* otmp  = (float*)take(BND * 4);
  float* amean = (float*)take(BNN * 4);
  float* atta  = (float*)take(BNN * 4);
  float* rowMv = (float*)take(BHN * 4);
  float* rowSv = (float*)take(BHN * 4);
  bf16*  hbf   = (bf16*)take(BND * 2);
  bf16*  xbf   = (bf16*)take(BND * 2);    // x bf16, later o bf16
  bf16*  wtbf  = (bf16*)take(DD * 2);
  bf16*  Qbf   = (bf16*)take(BND * 2);
  bf16*  Ktbf  = (bf16*)take(BND * 2);
  bf16*  Vbf   = (bf16*)take(BND * 2);
  bf16*  amnb  = (bf16*)take(BNN * 2);
  bf16*  attb  = (bf16*)take(BNN * 2);

  const dim3 blk256(256);
  const dim3 cgrid((unsigned)((BND + 255) / 256));
  const dim3 ngrid((unsigned)((BNN + 255) / 256));
  const dim3 tgrid((unsigned)((DD + 255) / 256));
  const dim3 gP(D_ / BN, (B_ * N_) / BM, 1);   // projections: 4 x 64
  const dim3 gC(N_ / BN, N_ / BM, B_);         // atten chain: 16 x 8 x 8
  const dim3 agrid(N_ / 16, H_, B_);
  const dim3 lngrid(B_ * N_);
  const dim3 blk32(32);

  // h = x @ W0^T  (bf16 mirror written for layer-0 projections)
  k_f32_to_bf16<<<cgrid, blk256, 0, stream>>>(x, xbf, BND);
  k_transpose_bf16<<<tgrid, blk256, 0, stream>>>(W0, wtbf, D_, D_);
  k_gemm_bf16<false, true><<<gP, blk256, 0, stream>>>(
      xbf, wtbf, hbuf, hbf, nullptr, D_, D_, 0, 0, 0);

  for (int L = 0; L < DEPTH_; ++L) {
    // Q / K / V projections
    k_transpose_bf16<<<tgrid, blk256, 0, stream>>>(Wq + L * DD, wtbf, D_, D_);
    k_gemm_bf16<true, false><<<gP, blk256, 0, stream>>>(
        hbf, wtbf, tmp, nullptr, bq + L * D_, D_, D_, 0, 0, 0);
    k_qkv_convert<0><<<cgrid, blk256, 0, stream>>>(tmp, Qbf);

    k_transpose_bf16<<<tgrid, blk256, 0, stream>>>(Wk + L * DD, wtbf, D_, D_);
    k_gemm_bf16<true, false><<<gP, blk256, 0, stream>>>(
        hbf, wtbf, tmp, nullptr, bk + L * D_, D_, D_, 0, 0, 0);
    k_qkv_convert<1><<<cgrid, blk256, 0, stream>>>(tmp, Ktbf);

    k_transpose_bf16<<<tgrid, blk256, 0, stream>>>(Wv + L * DD, wtbf, D_, D_);
    k_gemm_bf16<true, false><<<gP, blk256, 0, stream>>>(
        hbf, wtbf, tmp, nullptr, bv + L * D_, D_, D_, 0, 0, 0);
    k_qkv_convert<0><<<cgrid, blk256, 0, stream>>>(tmp, Vbf);

    // masked attention (two-pass flash-style) + head-averaged a_mean
    k_fill0<<<ngrid, blk256, 0, stream>>>(amean, BNN);
    k_attn_stats<<<agrid, blk32, 0, stream>>>(Qbf, Ktbf, adj, rowMv, rowSv);
    k_attn_apply<<<agrid, blk32, 0, stream>>>(Qbf, Ktbf, Vbf, adj,
                                              rowMv, rowSv, amean, otmp);

    // y = o @ Wo^T + bo
    k_f32_to_bf16<<<cgrid, blk256, 0, stream>>>(otmp, xbf, BND);
    k_transpose_bf16<<<tgrid, blk256, 0, stream>>>(Wo + L * DD, wtbf, D_, D_);
    k_gemm_bf16<true, false><<<gP, blk256, 0, stream>>>(
        xbf, wtbf, tmp, nullptr, bo + L * D_, D_, D_, 0, 0, 0);

    // h = LN(h + y)
    k_residual_ln<<<lngrid, blk32, 0, stream>>>(hbuf, tmp, gamma + L * D_,
                                                beta + L * D_, hbuf, hbf);

    // atten chain
    if (L == 0) {
      hipMemcpyAsync(atta, amean, BNN * 4, hipMemcpyDeviceToDevice, stream);
    } else {
      k_f32_to_bf16<<<ngrid, blk256, 0, stream>>>(amean, amnb, BNN);
      k_f32_to_bf16<<<ngrid, blk256, 0, stream>>>(atta, attb, BNN);
      k_gemm_bf16<false, false><<<gC, blk256, 0, stream>>>(
          amnb, attb, atta, nullptr, nullptr, N_, N_,
          (long long)N_ * N_, (long long)N_ * N_, (long long)N_ * N_);
    }
  }

  float* out = (float*)d_out;
  hipMemcpyAsync(out, hbuf, BND * 4, hipMemcpyDeviceToDevice, stream);
  hipMemcpyAsync(out + BND, atta, BNN * 4, hipMemcpyDeviceToDevice, stream);
}
